// Resnet_66924180406299
// MI455X (gfx1250) — matrix operand order repaired and checked by reading, not compiled, not run on hardware
//
#include <hip/hip_runtime.h>
#include <hip/hip_bf16.h>
#include <math.h>

// ---------------- problem constants ----------------
#define NB    64                   // graphs
#define NODES 52                   // nodes per graph
#define NN    (NB*NODES)           // 3328 nodes
#define LL    480                  // signal length after narrow(1,20,480)
#define CH    64                   // channels
static const long long PL = (long long)NN * LL;        // 1,597,440 positions
static const long long ACT = PL * CH;                  // 102,236,160 elements

typedef __bf16 bf16;
typedef __bf16 v16bf __attribute__((ext_vector_type(16)));
typedef __bf16 v8bf  __attribute__((ext_vector_type(8)));
typedef float  v8f   __attribute__((ext_vector_type(8)));

// =====================================================================
// small utility kernels
// =====================================================================
__global__ void k_zero(float* p, int n) {
    int i = blockIdx.x * blockDim.x + threadIdx.x;
    if (i < n) p[i] = 0.0f;
}

// BN statistics over bf16 activation [P rows][64 ch]
__global__ __launch_bounds__(256)
void k_bnstats_bf16(const bf16* __restrict__ src, long long P,
                    float* __restrict__ sum, float* __restrict__ sq) {
    int tid = threadIdx.x;
    int c   = tid & 63;
    int rs  = tid >> 6;                 // 0..3
    float s = 0.f, q = 0.f;
    for (long long r = (long long)blockIdx.x * 4 + rs; r < P; r += (long long)gridDim.x * 4) {
        float v = (float)src[r * CH + c];
        s += v; q += v * v;
    }
    __shared__ float S[256], Q[256];
    S[tid] = s; Q[tid] = q;
    __syncthreads();
    if (tid < 64) {
        float ts = S[tid] + S[tid + 64] + S[tid + 128] + S[tid + 192];
        float tq = Q[tid] + Q[tid + 64] + Q[tid + 128] + Q[tid + 192];
        atomicAdd(&sum[tid], ts);
        atomicAdd(&sq[tid],  tq);
    }
}

// BN statistics over f32 activation [P rows][64 ch]
__global__ __launch_bounds__(256)
void k_bnstats_f32(const float* __restrict__ src, long long P,
                   float* __restrict__ sum, float* __restrict__ sq) {
    int tid = threadIdx.x;
    int c   = tid & 63;
    int rs  = tid >> 6;
    float s = 0.f, q = 0.f;
    for (long long r = (long long)blockIdx.x * 4 + rs; r < P; r += (long long)gridDim.x * 4) {
        float v = src[r * CH + c];
        s += v; q += v * v;
    }
    __shared__ float S[256], Q[256];
    S[tid] = s; Q[tid] = q;
    __syncthreads();
    if (tid < 64) {
        atomicAdd(&sum[tid], S[tid] + S[tid + 64] + S[tid + 128] + S[tid + 192]);
        atomicAdd(&sq[tid],  Q[tid] + Q[tid + 64] + Q[tid + 128] + Q[tid + 192]);
    }
}

// fold mean/var + gamma/beta into per-channel scale/shift
__global__ void k_bnfinal(const float* __restrict__ sum, const float* __restrict__ sq,
                          float cnt, const float* __restrict__ g, const float* __restrict__ h,
                          float* __restrict__ sc, float* __restrict__ sh) {
    int c = threadIdx.x;
    float inv = 1.0f / cnt;
    float m = sum[c] * inv;
    float v = sq[c] * inv - m * m;
    float s = rsqrtf(v + 1e-5f) * g[c];
    sc[c] = s;
    sh[c] = h[c] - m * s;
}

// elementwise bf16: dst = [relu](src*sc + sh)
__global__ __launch_bounds__(256)
void k_bnapply_bf16(const bf16* __restrict__ src, bf16* __restrict__ dst,
                    const float* __restrict__ sc, const float* __restrict__ sh,
                    long long n, int relu) {
    long long i = (long long)blockIdx.x * blockDim.x + threadIdx.x;
    long long stride = (long long)gridDim.x * blockDim.x;
    for (; i < n; i += stride) {
        int c = (int)(i & 63);
        float v = (float)src[i] * sc[c] + sh[c];
        if (relu) v = fmaxf(v, 0.f);
        dst[i] = (bf16)v;
    }
}

// elementwise f32 in-place: v = [relu](v*sc + sh)
__global__ __launch_bounds__(256)
void k_bnapply_f32(float* __restrict__ t, const float* __restrict__ sc,
                   const float* __restrict__ sh, long long n, int relu) {
    long long i = (long long)blockIdx.x * blockDim.x + threadIdx.x;
    long long stride = (long long)gridDim.x * blockDim.x;
    for (; i < n; i += stride) {
        int c = (int)(i & 63);
        float v = t[i] * sc[c] + sh[c];
        if (relu) v = fmaxf(v, 0.f);
        t[i] = v;
    }
}

// dst = relu(a*scA+shA + b*scB+shB)   (residual join, bf16)
__global__ __launch_bounds__(256)
void k_combine(const bf16* __restrict__ a, const bf16* __restrict__ b, bf16* __restrict__ dst,
               const float* __restrict__ scA, const float* __restrict__ shA,
               const float* __restrict__ scB, const float* __restrict__ shB, long long n) {
    long long i = (long long)blockIdx.x * blockDim.x + threadIdx.x;
    long long stride = (long long)gridDim.x * blockDim.x;
    for (; i < n; i += stride) {
        int c = (int)(i & 63);
        float v = (float)a[i] * scA[c] + shA[c] + (float)b[i] * scB[c] + shB[c];
        dst[i] = (bf16)fmaxf(v, 0.f);
    }
}

// residual join + mean over L (block2 tail): pooled[n][c] = mean_l relu(...)
__global__ __launch_bounds__(256)
void k_combine_pool(const bf16* __restrict__ z, const bf16* __restrict__ s,
                    const float* __restrict__ scZ, const float* __restrict__ shZ,
                    const float* __restrict__ scS, const float* __restrict__ shS,
                    float* __restrict__ pooled) {
    int n = blockIdx.x;
    int tid = threadIdx.x;
    int c = tid & 63, ls = tid >> 6;
    float acc = 0.f;
    for (int l = ls; l < LL; l += 4) {
        long long idx = ((long long)n * LL + l) * CH + c;
        float v = (float)z[idx] * scZ[c] + shZ[c] + (float)s[idx] * scS[c] + shS[c];
        acc += fmaxf(v, 0.f);
    }
    __shared__ float red[256];
    red[tid] = acc;
    __syncthreads();
    if (tid < 64)
        pooled[(long long)n * CH + c] =
            (red[tid] + red[tid + 64] + red[tid + 128] + red[tid + 192]) * (1.0f / (float)LL);
}

// =====================================================================
// conv1d Cin=1 (block1 first conv K=7 and shortcut conv K=1): cheap VALU
// =====================================================================
__global__ __launch_bounds__(256)
void k_convc1(const float* __restrict__ x, const float* __restrict__ w,
              const float* __restrict__ bias, bf16* __restrict__ out,
              int K, int pad, long long total) {
    long long i = (long long)blockIdx.x * blockDim.x + threadIdx.x;
    long long stride = (long long)gridDim.x * blockDim.x;
    for (; i < total; i += stride) {
        int c = (int)(i & 63);
        long long p = i >> 6;
        int n = (int)(p / LL), l = (int)(p % LL);
        float acc = bias[c];
        for (int k = 0; k < K; ++k) {
            int r = l + k - pad;
            if (r >= 0 && r < LL) acc += w[c * K + k] * x[(long long)n * 500 + 20 + r];
        }
        out[i] = (bf16)acc;
    }
}

// =====================================================================
// weight repack for WMMA B fragments:
// packed[((tap*2+kk)*4 + cotile)*32 + lane][j] = w[co][ci][tap]
//   co = cotile*16 + (lane&15), ci = kk*32 + (j<8 ? (lane>>4)*8 + j : 16 + (lane>>4)*8 + j-8)
// =====================================================================
__global__ void k_pack_w(const float* __restrict__ w, int K, bf16* __restrict__ packed) {
    int total = K * 2 * 4 * 32 * 16;
    for (int idx = blockIdx.x * blockDim.x + threadIdx.x; idx < total;
         idx += gridDim.x * blockDim.x) {
        int j    = idx & 15;
        int lane = (idx >> 4) & 31;
        int t    = (idx >> 9) & 3;
        int kk   = (idx >> 11) & 1;
        int tap  = idx >> 12;
        int co = t * 16 + (lane & 15);
        int ci = kk * 32 + (j < 8 ? (lane >> 4) * 8 + j : 16 + (lane >> 4) * 8 + (j - 8));
        packed[idx] = (bf16)w[((long long)co * CH + ci) * K + tap];
    }
}

// =====================================================================
// 64->64 conv1d-as-GEMM on the bf16 WMMA pipe. Fully unrolled over taps.
// One wave computes a 32-position x 64-channel output tile:
//   two D(16x16 f32) tiles per co-tile; each B fragment feeds 2 WMMAs.
// =====================================================================
template <int K>
__global__ __launch_bounds__(128)
void k_conv_wmma(const bf16* __restrict__ in, const bf16* __restrict__ wpack,
                 const float* __restrict__ bias, bf16* __restrict__ out, int ntiles) {
    constexpr int PAD = (K - 1) / 2;
    int wave = blockIdx.x * (blockDim.x >> 5) + (threadIdx.x >> 5);
    if (wave >= ntiles) return;
    int lane = threadIdx.x & 31;
    int n  = wave / (LL / 32);            // 15 tiles of 32 positions per signal
    int l0 = (wave % (LL / 32)) * 32;
    int g = lane >> 4;                    // lane group
    int m = lane & 15;                    // row (A) / column (B,D) within group

    v8f acc[2][4];
#pragma unroll
    for (int u = 0; u < 2; ++u)
#pragma unroll
        for (int t = 0; t < 4; ++t) {
            float bv = bias[t * 16 + m];
#pragma unroll
            for (int v = 0; v < 8; ++v) acc[u][t][v] = bv;
        }

#pragma unroll
    for (int tap = 0; tap < K; ++tap) {
#pragma unroll
        for (int kk = 0; kk < 2; ++kk) {
            v16bf a[2];
#pragma unroll
            for (int u = 0; u < 2; ++u) {
                int row = l0 + u * 16 + m + tap - PAD;   // position of this lane's A row
                bool ok = (row >= 0) && (row < LL);
                if (ok) {
                    const bf16* rp = in + ((long long)n * LL + row) * CH;
                    // documented 16-bit A 16x32 layout:
                    // lanes0-15 K=0..7|16..23, lanes16-31 K=8..15|24..31
                    v8bf c0 = *reinterpret_cast<const v8bf*>(rp + kk * 32 + g * 8);
                    v8bf c1 = *reinterpret_cast<const v8bf*>(rp + kk * 32 + 16 + g * 8);
#pragma unroll
                    for (int j = 0; j < 8; ++j) { a[u][j] = c0[j]; a[u][8 + j] = c1[j]; }
                } else {
#pragma unroll
                    for (int j = 0; j < 16; ++j) a[u][j] = (bf16)0.0f;
                }
            }
#pragma unroll
            for (int t = 0; t < 4; ++t) {
                const bf16* bp = wpack + ((((tap * 2 + kk) * 4 + t) * 32 + lane) << 4);
                v16bf b = *reinterpret_cast<const v16bf*>(bp);
                acc[0][t] = __builtin_amdgcn_wmma_f32_16x16x32_bf16(
                    false, a[0], false, b, (short)0, acc[0][t], false, false);
                acc[1][t] = __builtin_amdgcn_wmma_f32_16x16x32_bf16(
                    false, a[1], false, b, (short)0, acc[1][t], false, false);
            }
        }
    }

#pragma unroll
    for (int u = 0; u < 2; ++u) {
        long long obase = ((long long)n * LL + l0 + u * 16) * CH;
#pragma unroll
        for (int t = 0; t < 4; ++t)
#pragma unroll
            for (int v = 0; v < 8; ++v) {
                int row = v + 8 * g;      // D layout: VGPR v -> M = v + 8*(lane>=16)
                out[obase + (long long)row * CH + t * 16 + m] = (bf16)acc[u][t][v];
            }
    }
}

// =====================================================================
// GAT pieces (dense per-graph attention; edges are complete + self loops)
// =====================================================================
__global__ __launch_bounds__(64)
void k_gat_h(const float* __restrict__ x, const float* __restrict__ W,
             const float* __restrict__ asrc, const float* __restrict__ adst,
             float* __restrict__ h, float* __restrict__ ss, float* __restrict__ sd) {
    int n = blockIdx.x, c = threadIdx.x;
    __shared__ float xr[64], r1[64], r2[64];
    xr[c] = x[(long long)n * CH + c];
    __syncthreads();
    float acc = 0.f;
    for (int k = 0; k < 64; ++k) acc += xr[k] * W[c * 64 + k];
    h[(long long)n * CH + c] = acc;
    r1[c] = acc * asrc[c];
    r2[c] = acc * adst[c];
    __syncthreads();
    for (int off = 32; off; off >>= 1) {
        if (c < off) { r1[c] += r1[c + off]; r2[c] += r2[c + off]; }
        __syncthreads();
    }
    if (c == 0) { ss[n] = r1[0]; sd[n] = r2[0]; }
}

__global__ __launch_bounds__(64)
void k_gat_attn(const float* __restrict__ h, const float* __restrict__ ss,
                const float* __restrict__ sd, const float* __restrict__ bias,
                float* __restrict__ out) {
    int i = blockIdx.x;                 // dst node
    int c = threadIdx.x;
    int base = (i / NODES) * NODES;
    __shared__ float e[NODES];
    if (c < NODES) {
        float v = ss[base + c] + sd[i];
        e[c] = (v > 0.f) ? v : 0.2f * v;   // LeakyReLU(0.2)
    }
    __syncthreads();
    float mx = -3.4e38f;
    for (int j = 0; j < NODES; ++j) mx = fmaxf(mx, e[j]);
    float den = 0.f;
    for (int j = 0; j < NODES; ++j) den += expf(e[j] - mx);
    float acc = 0.f;
    for (int j = 0; j < NODES; ++j)
        acc += expf(e[j] - mx) * h[(long long)(base + j) * CH + c];
    out[(long long)i * CH + c] = bias[c] + acc / den;
}

// global mean pool + argmax
__global__ __launch_bounds__(64)
void k_pool_out(const float* __restrict__ g, float* __restrict__ outp) {
    int b = blockIdx.x, c = threadIdx.x;
    float s = 0.f;
    for (int li = 0; li < NODES; ++li) s += g[(long long)(b * NODES + li) * CH + c];
    float v = s * (1.0f / NODES);
    outp[b * 64 + c] = v;
    __shared__ float vals[64];
    __shared__ int idxs[64];
    vals[c] = v; idxs[c] = c;
    __syncthreads();
    for (int off = 32; off; off >>= 1) {
        if (c < off) {
            if (vals[c + off] > vals[c]) { vals[c] = vals[c + off]; idxs[c] = idxs[c + off]; }
        }
        __syncthreads();
    }
    if (c == 0) outp[NB * 64 + b] = (float)idxs[0];
}

// =====================================================================
// host side
// =====================================================================
struct Params {
    const float *b1wx,*b1bx,*b1gx,*b1hx,*b1wy,*b1by,*b1gy,*b1hy,*b1wz,*b1bz,*b1gz,*b1hz,
                *b1gs,*b1hs,*b1ws,*b1bs;
    const float *b2wx,*b2bx,*b2gx,*b2hx,*b2wy,*b2by,*b2gy,*b2hy,*b2wz,*b2bz,*b2gz,*b2hz,
                *b2gs,*b2hs;
    const float *g1W,*g1as,*g1ad,*g1b,*g2W,*g2as,*g2ad,*g2b,*bn2g,*bn2b;
    const float *x;
};

extern "C" void kernel_launch(void* const* d_in, const int* in_sizes, int n_in,
                              void* d_out, int out_size, void* d_ws, size_t ws_size,
                              hipStream_t stream) {
    auto F = [&](int i) { return (const float*)d_in[i]; };
    Params p;
    if (in_sizes[0] == NN * 500) {
        // setup_inputs() insertion order: x, edge_index, batch, drop_rate, params(...)
        p.x = F(0);
        p.b1wx=F(4);  p.b1bx=F(5);  p.b1gx=F(6);  p.b1hx=F(7);
        p.b1wy=F(8);  p.b1by=F(9);  p.b1gy=F(10); p.b1hy=F(11);
        p.b1wz=F(12); p.b1bz=F(13); p.b1gz=F(14); p.b1hz=F(15);
        p.b1gs=F(16); p.b1hs=F(17); p.b1ws=F(18); p.b1bs=F(19);
        p.b2wx=F(20); p.b2bx=F(21); p.b2gx=F(22); p.b2hx=F(23);
        p.b2wy=F(24); p.b2by=F(25); p.b2gy=F(26); p.b2hy=F(27);
        p.b2wz=F(28); p.b2bz=F(29); p.b2gz=F(30); p.b2hz=F(31);
        p.b2gs=F(32); p.b2hs=F(33);
        p.g1W=F(34); p.g1as=F(35); p.g1ad=F(36); p.g1b=F(37);
        p.g2W=F(38); p.g2as=F(39); p.g2ad=F(40); p.g2b=F(41);
        p.bn2g=F(42); p.bn2b=F(43);
    } else {
        // jax tree-flatten alphabetical order
        p.b1bs=F(3);  p.b1bx=F(4);  p.b1by=F(5);  p.b1bz=F(6);
        p.b1gs=F(7);  p.b1gx=F(8);  p.b1gy=F(9);  p.b1gz=F(10);
        p.b1hs=F(11); p.b1hx=F(12); p.b1hy=F(13); p.b1hz=F(14);
        p.b1ws=F(15); p.b1wx=F(16); p.b1wy=F(17); p.b1wz=F(18);
        p.b2bx=F(19); p.b2by=F(20); p.b2bz=F(21);
        p.b2gs=F(22); p.b2gx=F(23); p.b2gy=F(24); p.b2gz=F(25);
        p.b2hs=F(26); p.b2hx=F(27); p.b2hy=F(28); p.b2hz=F(29);
        p.b2wx=F(30); p.b2wy=F(31); p.b2wz=F(32);
        p.bn2b=F(33); p.bn2g=F(34);
        p.g1W=F(35); p.g1ad=F(36); p.g1as=F(37); p.g1b=F(38);
        p.g2W=F(39); p.g2ad=F(40); p.g2as=F(41); p.g2b=F(42);
        p.x=F(43);
    }

    // -------- workspace carve --------
    char* w = (char*)d_ws;
    auto alloc = [&](size_t bytes) {
        char* r = w;
        w += ((bytes + 255) / 256) * 256;
        return (void*)r;
    };
    size_t actBytes = (size_t)ACT * sizeof(bf16);      // ~204.5 MB each
    bf16* t0 = (bf16*)alloc(actBytes);
    bf16* t1 = (bf16*)alloc(actBytes);
    bf16* t2 = (bf16*)alloc(actBytes);
    float* pooled = (float*)alloc((size_t)NN * 64 * 4);
    float* hbuf   = (float*)alloc((size_t)NN * 64 * 4);
    float* g1     = (float*)alloc((size_t)NN * 64 * 4);
    float* g2     = (float*)alloc((size_t)NN * 64 * 4);
    float* ssrc   = (float*)alloc((size_t)NN * 4);
    float* sdst   = (float*)alloc((size_t)NN * 4);
    float* stats  = (float*)alloc(128 * 4);            // sum[64] | sumsq[64]
    float* scA = (float*)alloc(64 * 4);
    float* shA = (float*)alloc(64 * 4);
    float* scB = (float*)alloc(64 * 4);
    float* shB = (float*)alloc(64 * 4);
    bf16* wpack = (bf16*)alloc((size_t)7 * 4096 * sizeof(bf16));
    (void)ws_size; (void)n_in; (void)out_size;

    const int EW_GRID = 99840;            // elementwise grid (4 elems/thread on ACT)
    const int NT32    = (int)(PL / 32);   // 49920 32-position WMMA tiles
    const int CONV_GRID = (NT32 + 3) / 4; // 4 waves per 128-thread block

    auto runBN_bf16 = [&](const bf16* t, long long P, const float* g, const float* h,
                          float* sc, float* sh) {
        k_zero<<<1, 128, 0, stream>>>(stats, 128);
        k_bnstats_bf16<<<1024, 256, 0, stream>>>(t, P, stats, stats + 64);
        k_bnfinal<<<1, 64, 0, stream>>>(stats, stats + 64, (float)P, g, h, sc, sh);
    };
    auto conv = [&](const bf16* in, const float* wt, const float* bias, bf16* out, int K) {
        k_pack_w<<<(K * 4096 + 255) / 256, 256, 0, stream>>>(wt, K, wpack);
        if (K == 7)
            k_conv_wmma<7><<<CONV_GRID, 128, 0, stream>>>(in, wpack, bias, out, NT32);
        else if (K == 5)
            k_conv_wmma<5><<<CONV_GRID, 128, 0, stream>>>(in, wpack, bias, out, NT32);
        else
            k_conv_wmma<3><<<CONV_GRID, 128, 0, stream>>>(in, wpack, bias, out, NT32);
    };

    // ---------------- block 1 (expand) ----------------
    k_convc1<<<EW_GRID, 256, 0, stream>>>(p.x, p.b1wx, p.b1bx, t0, 7, 3, ACT);
    runBN_bf16(t0, PL, p.b1gx, p.b1hx, scA, shA);
    k_bnapply_bf16<<<EW_GRID, 256, 0, stream>>>(t0, t1, scA, shA, ACT, 1);

    conv(t1, p.b1wy, p.b1by, t0, 5);
    runBN_bf16(t0, PL, p.b1gy, p.b1hy, scA, shA);
    k_bnapply_bf16<<<EW_GRID, 256, 0, stream>>>(t0, t0, scA, shA, ACT, 1);

    conv(t0, p.b1wz, p.b1bz, t1, 3);
    runBN_bf16(t1, PL, p.b1gz, p.b1hz, scA, shA);

    k_convc1<<<EW_GRID, 256, 0, stream>>>(p.x, p.b1ws, p.b1bs, t2, 1, 0, ACT);
    runBN_bf16(t2, PL, p.b1gs, p.b1hs, scB, shB);

    k_combine<<<EW_GRID, 256, 0, stream>>>(t1, t2, t0, scA, shA, scB, shB, ACT);  // h1 -> t0

    // ---------------- block 2 (identity shortcut) ----------------
    runBN_bf16(t0, PL, p.b2gs, p.b2hs, scB, shB);     // shortcut BN of block input

    conv(t0, p.b2wx, p.b2bx, t1, 7);
    runBN_bf16(t1, PL, p.b2gx, p.b2hx, scA, shA);
    k_bnapply_bf16<<<EW_GRID, 256, 0, stream>>>(t1, t1, scA, shA, ACT, 1);

    conv(t1, p.b2wy, p.b2by, t2, 5);
    runBN_bf16(t2, PL, p.b2gy, p.b2hy, scA, shA);
    k_bnapply_bf16<<<EW_GRID, 256, 0, stream>>>(t2, t2, scA, shA, ACT, 1);

    conv(t2, p.b2wz, p.b2bz, t1, 3);
    runBN_bf16(t1, PL, p.b2gz, p.b2hz, scA, shA);

    k_combine_pool<<<NN, 256, 0, stream>>>(t1, t0, scA, shA, scB, shB, pooled);   // [N,64]

    // ---------------- GAT 1 ----------------
    k_gat_h<<<NN, 64, 0, stream>>>(pooled, p.g1W, p.g1as, p.g1ad, hbuf, ssrc, sdst);
    k_gat_attn<<<NN, 64, 0, stream>>>(hbuf, ssrc, sdst, p.g1b, g1);

    // bn2 + relu (per channel over all N nodes)
    k_zero<<<1, 128, 0, stream>>>(stats, 128);
    k_bnstats_f32<<<256, 256, 0, stream>>>(g1, NN, stats, stats + 64);
    k_bnfinal<<<1, 64, 0, stream>>>(stats, stats + 64, (float)NN, p.bn2g, p.bn2b, scA, shA);
    k_bnapply_f32<<<(NN * 64 + 255) / 256, 256, 0, stream>>>(g1, scA, shA, (long long)NN * 64, 1);

    // ---------------- GAT 2 + pooling + argmax ----------------
    k_gat_h<<<NN, 64, 0, stream>>>(g1, p.g2W, p.g2as, p.g2ad, hbuf, ssrc, sdst);
    k_gat_attn<<<NN, 64, 0, stream>>>(hbuf, ssrc, sdst, p.g2b, g2);
    k_pool_out<<<NB, 64, 0, stream>>>(g2, (float*)d_out);
}
